// Seq2seq_30528627540577
// MI455X (gfx1250) — matrix operand order repaired and checked by reading, not compiled, not run on hardware
//
#include <hip/hip_runtime.h>
#include <hip/hip_bf16.h>
#include <math.h>

typedef __attribute__((ext_vector_type(16))) _Float16 v16h;
typedef __attribute__((ext_vector_type(8)))  _Float16 v8h;
typedef __attribute__((ext_vector_type(8)))  float    v8f;

#define B_   64
#define S_   40
#define T_   39          // decoder steps = T-1
#define E_   256
#define U_   1024
#define G3_  3072        // 3*U
#define KD_  1280        // U+E (decoder gx K)
#define VO_  8192

// ---------------- utility kernels ----------------

__global__ void k_zero_h(float* h32, _Float16* h16, int n) {
    int i = blockIdx.x * blockDim.x + threadIdx.x;
    if (i < n) { h32[i] = 0.f; h16[i] = (_Float16)0.f; }
}

// in: (R x C) f32 row-major  ->  out: (C x R) f16 row-major (so each out-row is K-contiguous)
__global__ void k_transpose_cvt(const float* __restrict__ in, _Float16* __restrict__ out,
                                int R, int C) {
    long i = (long)blockIdx.x * blockDim.x + threadIdx.x;
    long total = (long)R * C;
    if (i < total) {
        int  c = (int)(i % C);
        long r = i / C;
        out[(long)c * R + r] = (_Float16)in[i];
    }
}

__global__ void k_embed_enc(const int* __restrict__ inp, const float* __restrict__ emb,
                            _Float16* __restrict__ xemb) {
    int i = blockIdx.x * blockDim.x + threadIdx.x;           // B*S*E
    if (i < B_ * S_ * E_) {
        int e = i % E_;
        int r = i / E_;                                      // r = b*S + s
        int tok = inp[r];
        xemb[i] = (_Float16)emb[(long)tok * E_ + e];
    }
}

// fill XIN[t][b][U_ + e] with dec_emb[targ[b][t]]  (targ row stride = S_ = 40, use t<39)
__global__ void k_embed_dec(const int* __restrict__ targ, const float* __restrict__ emb,
                            _Float16* __restrict__ xin) {
    int i = blockIdx.x * blockDim.x + threadIdx.x;           // B*T*E
    if (i < B_ * T_ * E_) {
        int e = i % E_;
        int r = i / E_;
        int t = r % T_;
        int b = r / T_;
        int tok = targ[b * S_ + t];
        xin[((long)t * B_ + b) * KD_ + U_ + e] = (_Float16)emb[(long)tok * E_ + e];
    }
}

// ---------------- WMMA GEMM ----------------
// C(M x N) = A(M x K, f16 row-major, lda) * Bt(N x K, f16, each row = one column's K) + bias[col]
// grid: (N/128, M/16), block: 256 (8 waves, 1 wave per 16x16 tile). M%16==0, N%128==0, K%32==0.
__global__ void k_wmma_gemm(const _Float16* __restrict__ A, int lda,
                            const _Float16* __restrict__ Bt,
                            const float* __restrict__ bias,
                            float* __restrict__ C, long ldc, int K) {
    const int lane = threadIdx.x & 31;
    const int wave = threadIdx.x >> 5;
    const int hh   = lane >> 4;       // half-wave select
    const int l15  = lane & 15;
    const int row0 = blockIdx.y * 16;
    const int col  = (blockIdx.x * 8 + wave) * 16 + l15;

    const _Float16* arow = A  + (long)(row0 + l15) * lda;
    const _Float16* bcol = Bt + (long)col * K;

    v8f c = {};
    for (int k0 = 0; k0 < K; k0 += 32) {
        // A fragment: lane holds row (lane&15), K = [k0+16*hh, +16) contiguous
        v16h a = *(const v16h*)(arow + k0 + 16 * hh);
        // B fragment: lane holds col (lane&15), K = [k0+16*hh, +16) contiguous
        v16h b = *(const v16h*)(bcol + k0 + 16 * hh);
        c = __builtin_amdgcn_wmma_f32_16x16x32_f16(false, a, false, b, (short)0, c,
                                                   false, false);
    }
    const float bv = bias ? bias[col] : 0.f;
#pragma unroll
    for (int r = 0; r < 8; ++r) {
        C[(long)(row0 + r + 8 * hh) * ldc + col] = c[r] + bv;
    }
}

// ---------------- GRU pointwise ----------------

__device__ __forceinline__ float sigf(float x) { return 1.f / (1.f + expf(-x)); }

// encoder step: h = z*h + (1-z)*c ; write h (f32+f16) and enc_out[b][s][:]
__global__ void k_gru_enc(const float* __restrict__ gx,   // (B*S x 3072), bias0 included
                          const float* __restrict__ gh,   // (B x 3072), bias1 included
                          float* __restrict__ h32, _Float16* __restrict__ h16,
                          float* __restrict__ enc_out, int s) {
    int i = blockIdx.x * blockDim.x + threadIdx.x;         // B*U
    if (i >= B_ * U_) return;
    int j = i % U_, b = i / U_;
    const float* gxr = gx + (long)(b * S_ + s) * G3_;
    const float* ghr = gh + (long)b * G3_;
    float z  = sigf(gxr[j] + ghr[j]);
    float r  = sigf(gxr[U_ + j] + ghr[U_ + j]);
    float cc = tanhf(gxr[2 * U_ + j] + r * ghr[2 * U_ + j]);
    float hn = z * h32[i] + (1.f - z) * cc;
    h32[i] = hn;
    h16[i] = (_Float16)hn;
    enc_out[((long)b * S_ + s) * U_ + j] = hn;
}

// decoder step: h_prev == 0 in reference, so gh = dec_b[1] and h_new = (1-z)*c
__global__ void k_gru_dec(const float* __restrict__ g,    // (B x 3072), bias0 included
                          const float* __restrict__ bh,   // dec_b row 1 (3072)
                          float* __restrict__ h32, _Float16* __restrict__ h16) {
    int i = blockIdx.x * blockDim.x + threadIdx.x;         // B*U
    if (i >= B_ * U_) return;
    int j = i % U_, b = i / U_;
    const float* gr = g + (long)b * G3_;
    float z  = sigf(gr[j] + bh[j]);
    float r  = sigf(gr[U_ + j] + bh[U_ + j]);
    float cc = tanhf(gr[2 * U_ + j] + r * bh[2 * U_ + j]);
    float hn = (1.f - z) * cc;
    h32[i] = hn;
    h16[i] = (_Float16)hn;
}

// ---------------- attention (per decoder step) ----------------
// block per batch element; scores = h . enc_out[b,s,:], softmax, ctx -> XIN[t][b][0:U] (f16)
__global__ void k_attention(const float* __restrict__ h32,
                            const float* __restrict__ enc_out,
                            _Float16* __restrict__ xin, int t) {
    const int b    = blockIdx.x;
    const int tid  = threadIdx.x;
    const int lane = tid & 31;
    const int wave = tid >> 5;                              // 8 waves
    __shared__ float sc[S_];
    __shared__ float s_inv;
    const float* hb = h32 + (long)b * U_;
    const float* eb = enc_out + (long)b * S_ * U_;

    for (int s = wave; s < S_; s += 8) {
        const float* er = eb + (long)s * U_;
        float acc = 0.f;
        for (int u = lane; u < U_; u += 32) acc += hb[u] * er[u];
        for (int off = 16; off > 0; off >>= 1) acc += __shfl_down(acc, off, 32);
        if (lane == 0) sc[s] = acc;
    }
    __syncthreads();
    if (tid == 0) {
        float mx = sc[0];
        for (int s = 1; s < S_; ++s) mx = fmaxf(mx, sc[s]);
        float sum = 0.f;
        for (int s = 0; s < S_; ++s) { float e = expf(sc[s] - mx); sc[s] = e; sum += e; }
        s_inv = 1.f / sum;
    }
    __syncthreads();
    const float inv = s_inv;
    _Float16* xr = xin + ((long)t * B_ + b) * KD_;
    for (int j = tid; j < U_; j += blockDim.x) {
        float acc = 0.f;
        for (int s = 0; s < S_; ++s) acc += sc[s] * eb[(long)s * U_ + j];
        xr[j] = (_Float16)(acc * inv);
    }
}

// ---------------- host launcher ----------------

extern "C" void kernel_launch(void* const* d_in, const int* in_sizes, int n_in,
                              void* d_out, int out_size, void* d_ws, size_t ws_size,
                              hipStream_t stream) {
    const int*   inp     = (const int*)  d_in[0];
    const int*   targ    = (const int*)  d_in[1];
    const float* enc_emb = (const float*)d_in[2];
    const float* enc_Wx  = (const float*)d_in[3];
    const float* enc_Wh  = (const float*)d_in[4];
    const float* enc_b   = (const float*)d_in[5];
    const float* dec_emb = (const float*)d_in[6];
    const float* dec_Wx  = (const float*)d_in[7];
    /* d_in[8] = dec_Wh: multiplied by h_zero in the reference -> unused */
    const float* dec_b   = (const float*)d_in[9];
    const float* fc_W    = (const float*)d_in[10];
    const float* fc_b    = (const float*)d_in[11];
    float* out = (float*)d_out;

    char* p = (char*)d_ws;
    auto alloc = [&](size_t bytes) -> void* {
        void* r = (void*)p;
        p += (bytes + 255) & ~(size_t)255;
        return r;
    };
    _Float16* encWxT = (_Float16*)alloc(sizeof(_Float16) * (size_t)G3_ * E_);
    _Float16* encWhT = (_Float16*)alloc(sizeof(_Float16) * (size_t)G3_ * U_);
    _Float16* decWxT = (_Float16*)alloc(sizeof(_Float16) * (size_t)G3_ * KD_);
    _Float16* fcWT   = (_Float16*)alloc(sizeof(_Float16) * (size_t)VO_ * U_);
    _Float16* xemb   = (_Float16*)alloc(sizeof(_Float16) * (size_t)B_ * S_ * E_);
    float*    GXenc  = (float*)   alloc(sizeof(float)    * (size_t)B_ * S_ * G3_);
    float*    encOut = (float*)   alloc(sizeof(float)    * (size_t)B_ * S_ * U_);
    float*    h32    = (float*)   alloc(sizeof(float)    * (size_t)B_ * U_);
    _Float16* h16    = (_Float16*)alloc(sizeof(_Float16) * (size_t)B_ * U_);
    float*    ghbuf  = (float*)   alloc(sizeof(float)    * (size_t)B_ * G3_);
    _Float16* XIN    = (_Float16*)alloc(sizeof(_Float16) * (size_t)T_ * B_ * KD_);
    float*    gdec   = (float*)   alloc(sizeof(float)    * (size_t)B_ * G3_);
    (void)ws_size; (void)n_in; (void)in_sizes; (void)out_size;

    // weight convert+transpose to f16 (N x K)
    {
        long n;
        n = (long)E_  * G3_; k_transpose_cvt<<<(n + 255) / 256, 256, 0, stream>>>(enc_Wx, encWxT, E_,  G3_);
        n = (long)U_  * G3_; k_transpose_cvt<<<(n + 255) / 256, 256, 0, stream>>>(enc_Wh, encWhT, U_,  G3_);
        n = (long)KD_ * G3_; k_transpose_cvt<<<(n + 255) / 256, 256, 0, stream>>>(dec_Wx, decWxT, KD_, G3_);
        n = (long)U_  * VO_; k_transpose_cvt<<<(n + 255) / 256, 256, 0, stream>>>(fc_W,   fcWT,   U_,  VO_);
    }

    k_zero_h<<<(B_ * U_ + 255) / 256, 256, 0, stream>>>(h32, h16, B_ * U_);
    k_embed_enc<<<(B_ * S_ * E_ + 255) / 256, 256, 0, stream>>>(inp, enc_emb, xemb);
    k_embed_dec<<<(B_ * T_ * E_ + 255) / 256, 256, 0, stream>>>(targ, dec_emb, XIN);

    // all-step encoder input projection: GXenc = xemb(2560x256) @ Wx + b0
    k_wmma_gemm<<<dim3(G3_ / 128, (B_ * S_) / 16), 256, 0, stream>>>(
        xemb, E_, encWxT, enc_b, GXenc, G3_, E_);

    // encoder recurrence
    for (int s = 0; s < S_; ++s) {
        k_wmma_gemm<<<dim3(G3_ / 128, B_ / 16), 256, 0, stream>>>(
            h16, U_, encWhT, enc_b + G3_, ghbuf, G3_, U_);
        k_gru_enc<<<(B_ * U_ + 255) / 256, 256, 0, stream>>>(
            GXenc, ghbuf, h32, h16, encOut, s);
    }

    // decoder (carry h starts at encoder final state, already in h32/h16)
    for (int t = 0; t < T_; ++t) {
        k_attention<<<B_, 256, 0, stream>>>(h32, encOut, XIN, t);
        k_wmma_gemm<<<dim3(G3_ / 128, B_ / 16), 256, 0, stream>>>(
            XIN + (size_t)t * B_ * KD_, KD_, decWxT, dec_b, gdec, G3_, KD_);
        k_gru_dec<<<(B_ * U_ + 255) / 256, 256, 0, stream>>>(
            gdec, dec_b + G3_, h32, h16);
        k_wmma_gemm<<<dim3(VO_ / 128, B_ / 16), 256, 0, stream>>>(
            h16, U_, fcWT, fc_b, out + (size_t)t * VO_, (long)T_ * VO_, U_);
    }
}
